// JobActor_7404523618592
// MI455X (gfx1250) — matrix operand order repaired and checked by reading, not compiled, not run on hardware
//
#include <hip/hip_runtime.h>
#include <math.h>

// ---------- vector types (POD, union/shuffle-safe) ----------
typedef _Float16 v16h __attribute__((ext_vector_type(16)));
typedef _Float16 h8   __attribute__((ext_vector_type(8)));
typedef _Float16 h4   __attribute__((ext_vector_type(4)));
typedef float    v8f  __attribute__((ext_vector_type(8)));
typedef float    f4   __attribute__((ext_vector_type(4)));

// =====================================================================
// Weight packing: row-major f32 W[K][M] -> WMMA-B-fragment-ordered f16.
// Fragment layout (V_WMMA_F32_16X16X32_F16, wave32):
//   lane n (0..15):  col = ct*16 + n, halves h=0..15 hold K = ks*32 + {0..7, 16..23}
//   lane n (16..31): col = ct*16 + (n-16), halves hold K = ks*32 + {8..15, 24..31}
// Packed index: ((ks*nct + ct)*32 + lane)*16 + h   (contiguous 32B per lane)
// =====================================================================
__global__ void pack_weight(const float* __restrict__ W, _Float16* __restrict__ out,
                            int K, int M) {
  int idx = blockIdx.x * blockDim.x + threadIdx.x;
  int total = K * M;
  if (idx >= total) return;
  int h    = idx & 15;
  int lane = (idx >> 4) & 31;
  int rest = idx >> 9;
  int nct  = M >> 4;
  int ct   = rest % nct;
  int ks   = rest / nct;
  int k    = ks * 32 + ((h >> 3) << 4) + ((lane >> 4) << 3) + (h & 7);
  int m    = ct * 16 + (lane & 15);
  out[idx] = (_Float16)W[(size_t)k * M + m];
}

// =====================================================================
// WMMA GEMM: C[Npad x M](f16) = act( A[Npad x K](f16) @ Bpacked + bias )
// 256 threads = 8 waves; block = one 16-row tile; wave owns col-tiles.
// A fragment (16x32 f16): lane<16 -> row=lane, K={k0..k0+7, k0+16..k0+23}
//                         lane>=16 -> row=lane-16, K offset +8.
// =====================================================================
template<bool RELU>
__global__ __launch_bounds__(256) void gemm16(const _Float16* __restrict__ A,
                                              const _Float16* __restrict__ Bp,
                                              const float*    __restrict__ bias,
                                              _Float16*       __restrict__ C,
                                              int K, int M) {
  const int lane = threadIdx.x & 31;
  const int wave = threadIdx.x >> 5;
  const int rt   = blockIdx.x;
  const int nct  = M >> 4;
  const int nks  = K >> 5;
  const int row  = (rt << 4) + (lane & 15);
  const int koff = (lane >> 4) << 3;
  const _Float16* arow = A + (size_t)row * K + koff;

  for (int ct = wave; ct < nct; ct += 8) {
    v8f acc = {};
    const _Float16* bp = Bp + (((size_t)ct << 5) + lane) * 16;
#pragma unroll 4
    for (int ks = 0; ks < nks; ++ks) {
      h8 alo = *(const h8*)(arow + (ks << 5));
      h8 ahi = *(const h8*)(arow + (ks << 5) + 16);
      v16h av = __builtin_shufflevector(alo, ahi,
                 0,1,2,3,4,5,6,7,8,9,10,11,12,13,14,15);
      const h8* bq = (const h8*)(bp + (size_t)ks * nct * 512);
      h8 blo = bq[0];
      h8 bhi = bq[1];
      v16h bv = __builtin_shufflevector(blo, bhi,
                 0,1,2,3,4,5,6,7,8,9,10,11,12,13,14,15);
      acc = __builtin_amdgcn_wmma_f32_16x16x32_f16(
              false, av, false, bv, (short)0, acc, false, false);
    }
    const int col = (ct << 4) + (lane & 15);
    const float bv = bias[col];
    const int r0 = (rt << 4) + ((lane >> 4) << 3);
#pragma unroll
    for (int r = 0; r < 8; ++r) {
      float v = acc[r] + bv;
      if (RELU) v = v > 0.f ? v : 0.f;
      C[(size_t)(r0 + r) * M + col] = (_Float16)v;
    }
  }
}

// ---------- misc elementwise / graph kernels ----------
__global__ void zero_f32(float* p, int n) {
  int i = blockIdx.x * blockDim.x + threadIdx.x;
  if (i < n) p[i] = 0.f;
}

__global__ void feat_init(const float* __restrict__ tss, const float* __restrict__ tsr,
                          const float* __restrict__ tct, float* __restrict__ x0,
                          float* __restrict__ agg3, int N) {
  int i = blockIdx.x * blockDim.x + threadIdx.x;
  if (i >= N) return;
  x0[i*3+0] = tss[i]; x0[i*3+1] = tsr[i]; x0[i*3+2] = tct[i];
  agg3[i*3+0] = 0.f;  agg3[i*3+1] = 0.f;  agg3[i*3+2] = 0.f;
}

__global__ void scatter3(const float* __restrict__ x0, const int* __restrict__ src,
                         const int* __restrict__ dst, float* __restrict__ agg3, int E) {
  int e = blockIdx.x * blockDim.x + threadIdx.x;
  if (e >= E) return;
  int s = src[e], d = dst[e];
  atomicAdd(&agg3[d*3+0], x0[s*3+0]);
  atomicAdd(&agg3[d*3+1], x0[s*3+1]);
  atomicAdd(&agg3[d*3+2], x0[s*3+2]);
}

// GIN layer 0 first linear (K=3): t = relu((x0+agg3) @ W1(3x128) + b1)
__global__ void gin0_mlp1(const float* __restrict__ x0, const float* __restrict__ agg3,
                          const float* __restrict__ W1, const float* __restrict__ b1,
                          _Float16* __restrict__ out, int N) {
  int t = blockIdx.x * blockDim.x + threadIdx.x;
  int i = t >> 7, j = t & 127;
  if (i >= N) return;
  float z0 = x0[i*3+0] + agg3[i*3+0];
  float z1 = x0[i*3+1] + agg3[i*3+1];
  float z2 = x0[i*3+2] + agg3[i*3+2];
  float a = b1[j] + z0 * W1[j] + z1 * W1[128 + j] + z2 * W1[256 + j];
  out[(size_t)i * 128 + j] = (_Float16)fmaxf(a, 0.f);
}

// edge scatter for 128-dim features: agg[dst] += x[src]  (L2-resident atomics)
__global__ void scatter128(const _Float16* __restrict__ x, const int* __restrict__ src,
                           const int* __restrict__ dst, float* __restrict__ agg, int E) {
  long long tid = (long long)blockIdx.x * blockDim.x + threadIdx.x;
  int e = (int)(tid >> 5);
  if (e >= E) return;
  int c = ((int)tid & 31) << 2;
  int s = src[e], d = dst[e];
  h4 v = *(const h4*)(x + (size_t)s * 128 + c);
  float* a = agg + (size_t)d * 128 + c;
  atomicAdd(a + 0, (float)v[0]);
  atomicAdd(a + 1, (float)v[1]);
  atomicAdd(a + 2, (float)v[2]);
  atomicAdd(a + 3, (float)v[3]);
}

// z = (f16)( (f32)x + agg )
__global__ void add_cvt(const _Float16* __restrict__ x, const float* __restrict__ agg,
                        _Float16* __restrict__ z, int N) {
  int t = blockIdx.x * blockDim.x + threadIdx.x;
  if (t >= N * 32) return;
  size_t base = (size_t)t * 4;
  h4 xv = *(const h4*)(x + base);
  f4 av = *(const f4*)(agg + base);
  h4 zv;
  zv[0] = (_Float16)((float)xv[0] + av[0]);
  zv[1] = (_Float16)((float)xv[1] + av[1]);
  zv[2] = (_Float16)((float)xv[2] + av[2]);
  zv[3] = (_Float16)((float)xv[3] + av[3]);
  *(h4*)(z + base) = zv;
}

// column sums over N rows of x (f16, 128 cols) -> gsum (pre-zeroed)
__global__ void colsum(const _Float16* __restrict__ x, float* __restrict__ gsum, int N) {
  float acc = 0.f;
  for (int i = blockIdx.x; i < N; i += gridDim.x)
    acc += (float)x[(size_t)i * 128 + threadIdx.x];
  atomicAdd(&gsum[threadIdx.x], acc);
}

__global__ void finalize_gemb(const float* __restrict__ gsum, float* __restrict__ gemb,
                              float* __restrict__ d_out, int N) {
  int j = threadIdx.x;
  float g = gsum[j] / (float)N;
  gemb[j] = g;
  d_out[N + j] = g;
}

// machine MLP: memb = relu(vct @ mW1 + mb1) @ mW2 + mb2   (1x64 -> 1x128)
__global__ void machine_mlp(const float* __restrict__ vct,
                            const float* __restrict__ W1, const float* __restrict__ b1,
                            const float* __restrict__ W2, const float* __restrict__ b2,
                            float* __restrict__ memb, float* __restrict__ d_out, int N) {
  __shared__ float hm[128];
  int j = threadIdx.x;
  float a = b1[j];
  for (int k = 0; k < 64; ++k) a += vct[k] * W1[k * 128 + j];
  hm[j] = fmaxf(a, 0.f);
  __syncthreads();
  float e = b2[j];
  for (int k = 0; k < 128; ++k) e += hm[k] * W2[k * 128 + j];
  memb[j] = e;
  d_out[N + 128 + j] = e;
}

// effbias[j] = d_b1[j] + g . d_W1[128:256, j] + m . d_W1[256:384, j]
__global__ void eff_bias(const float* __restrict__ gemb, const float* __restrict__ memb,
                         const float* __restrict__ dW1, const float* __restrict__ db1,
                         float* __restrict__ eff) {
  int j = threadIdx.x;
  float a = db1[j];
  for (int k = 0; k < 128; ++k) a += gemb[k] * dW1[(size_t)(128 + k) * 256 + j];
  for (int k = 0; k < 128; ++k) a += memb[k] * dW1[(size_t)(256 + k) * 256 + j];
  eff[j] = a;
}

__global__ void score_kernel(const _Float16* __restrict__ h2, const float* __restrict__ W3,
                             const float* __restrict__ b3, const float* __restrict__ tsr,
                             float* __restrict__ scores, int N) {
  int i = blockIdx.x * blockDim.x + threadIdx.x;
  if (i >= N) return;
  float s = b3[0];
  const _Float16* row = h2 + (size_t)i * 128;
  for (int j = 0; j < 128; ++j) s += (float)row[j] * W3[j];
  if (tsr[i] == 0.f) s = -INFINITY;
  scores[i] = s;
}

// ---------- softmax reductions (deterministic 5-pass) ----------
__global__ void part_max(const float* __restrict__ s, float* __restrict__ part, int N) {
  __shared__ float sm[256];
  float m = -INFINITY;
  for (int i = blockIdx.x * 256 + threadIdx.x; i < N; i += gridDim.x * 256)
    m = fmaxf(m, s[i]);
  sm[threadIdx.x] = m; __syncthreads();
  for (int o = 128; o > 0; o >>= 1) {
    if (threadIdx.x < o) sm[threadIdx.x] = fmaxf(sm[threadIdx.x], sm[threadIdx.x + o]);
    __syncthreads();
  }
  if (threadIdx.x == 0) part[blockIdx.x] = sm[0];
}

__global__ void final_max(const float* __restrict__ part, float* __restrict__ gms) {
  __shared__ float sm[256];
  sm[threadIdx.x] = part[threadIdx.x]; __syncthreads();
  for (int o = 128; o > 0; o >>= 1) {
    if (threadIdx.x < o) sm[threadIdx.x] = fmaxf(sm[threadIdx.x], sm[threadIdx.x + o]);
    __syncthreads();
  }
  if (threadIdx.x == 0) gms[0] = sm[0];
}

__global__ void part_sum(const float* __restrict__ s, const float* __restrict__ gms,
                         float* __restrict__ part, int N) {
  __shared__ float sm[256];
  float gmax = gms[0];
  float a = 0.f;
  for (int i = blockIdx.x * 256 + threadIdx.x; i < N; i += gridDim.x * 256)
    a += expf(s[i] - gmax);
  sm[threadIdx.x] = a; __syncthreads();
  for (int o = 128; o > 0; o >>= 1) {
    if (threadIdx.x < o) sm[threadIdx.x] += sm[threadIdx.x + o];
    __syncthreads();
  }
  if (threadIdx.x == 0) part[blockIdx.x] = sm[0];
}

__global__ void final_sum(const float* __restrict__ part, float* __restrict__ gms) {
  __shared__ float sm[256];
  sm[threadIdx.x] = part[threadIdx.x]; __syncthreads();
  for (int o = 128; o > 0; o >>= 1) {
    if (threadIdx.x < o) sm[threadIdx.x] += sm[threadIdx.x + o];
    __syncthreads();
  }
  if (threadIdx.x == 0) gms[1] = sm[0];
}

__global__ void probs_kernel(const float* __restrict__ s, const float* __restrict__ gms,
                             float* __restrict__ out, int N) {
  int i = blockIdx.x * blockDim.x + threadIdx.x;
  if (i >= N) return;
  out[i] = expf(s[i] - gms[0]) / gms[1];
}

// =====================================================================
// Host-side orchestration
// =====================================================================
extern "C" void kernel_launch(void* const* d_in, const int* in_sizes, int n_in,
                              void* d_out, int out_size, void* d_ws, size_t ws_size,
                              hipStream_t stream) {
  (void)out_size; (void)ws_size;
  const float* tss = (const float*)d_in[0];
  const float* tsr = (const float*)d_in[1];
  const float* tct = (const float*)d_in[2];
  const float* vct = (const float*)d_in[3];
  const int N    = in_sizes[0];
  const int Npad = (N + 15) & ~15;

  const float *dW1,*dW2,*dW3,*db1,*db2,*db3;
  const float *g0W1,*g0W2,*g0b1,*g0b2,*g1W1,*g1W2,*g1b1,*g1b2,*g2W1,*g2W2,*g2b1,*g2b2;
  const float *mW1,*mW2,*mb1,*mb2;
  const int* edges;

  if (n_in == 6) {
    // params flattened into one buffer, jax pytree (sorted-key) order
    const float* P = (const float*)d_in[4];
    size_t off = 0;
    auto nxt = [&](size_t n) { const float* p = P + off; off += n; return p; };
    dW1 = nxt(384*256); dW2 = nxt(256*128); dW3 = nxt(128);
    db1 = nxt(256);     db2 = nxt(128);     db3 = nxt(1);
    g0W1 = nxt(3*128);   g0W2 = nxt(128*128); g0b1 = nxt(128); g0b2 = nxt(128);
    g1W1 = nxt(128*128); g1W2 = nxt(128*128); g1b1 = nxt(128); g1b2 = nxt(128);
    g2W1 = nxt(128*128); g2W2 = nxt(128*128); g2b1 = nxt(128); g2b2 = nxt(128);
    mW1 = nxt(64*128);   mW2 = nxt(128*128);  mb1 = nxt(128);  mb2 = nxt(128);
    edges = (const int*)d_in[5];
  } else {
    // params expanded into separate leaves
    const float* const* q = (const float* const*)(d_in + 4);
    if (in_sizes[4] == 384*256) { // sorted-key order
      dW1=q[0]; dW2=q[1]; dW3=q[2]; db1=q[3]; db2=q[4]; db3=q[5];
      g0W1=q[6];  g0W2=q[7];  g0b1=q[8];  g0b2=q[9];
      g1W1=q[10]; g1W2=q[11]; g1b1=q[12]; g1b2=q[13];
      g2W1=q[14]; g2W2=q[15]; g2b1=q[16]; g2b2=q[17];
      mW1=q[18]; mW2=q[19]; mb1=q[20]; mb2=q[21];
    } else {                       // insertion order
      g0W1=q[0];  g0b1=q[1];  g0W2=q[2];  g0b2=q[3];
      g1W1=q[4];  g1b1=q[5];  g1W2=q[6];  g1b2=q[7];
      g2W1=q[8];  g2b1=q[9];  g2W2=q[10]; g2b2=q[11];
      mW1=q[12]; mb1=q[13]; mW2=q[14]; mb2=q[15];
      dW1=q[16]; db1=q[17]; dW2=q[18]; db2=q[19]; dW3=q[20]; db3=q[21];
    }
    edges = (const int*)d_in[n_in - 1];
  }
  const int E = in_sizes[n_in - 1] / 2;
  const int* esrc = edges;
  const int* edst = edges + E;

  // ---------- workspace layout ----------
  char* wsb = (char*)d_ws;
  size_t off = 0;
  auto take = [&](size_t bytes) -> char* {
    char* p = wsb + off; off = (off + bytes + 255) & ~(size_t)255; return p;
  };
  _Float16* xf16 = (_Float16*)take((size_t)Npad * 128 * 2);           // node features
  float*    agg  = (float*)   take((size_t)Npad * 256 * 2);           // f32 agg / h1 f16
  _Float16* zf16 = (_Float16*)take((size_t)Npad * 128 * 2);           // z / h2
  _Float16* tf16 = (_Float16*)take((size_t)Npad * 128 * 2);           // hidden t
  float*    x0   = (float*)   take((size_t)N * 3 * 4);
  float*    agg3 = (float*)   take((size_t)N * 3 * 4);
  float*    scr  = (float*)   take((size_t)N * 4);
  _Float16* pw   = (_Float16*)take((size_t)(5 * 16384 + 2 * 32768) * 2);
  float*    sm   = (float*)   take(8192);

  _Float16* p_g0W2 = pw;
  _Float16* p_g1W1 = p_g0W2 + 16384;
  _Float16* p_g1W2 = p_g1W1 + 16384;
  _Float16* p_g2W1 = p_g1W2 + 16384;
  _Float16* p_g2W2 = p_g2W1 + 16384;
  _Float16* p_d1   = p_g2W2 + 16384;   // 128x256
  _Float16* p_d2   = p_d1   + 32768;   // 256x128

  float* gsum = sm;            // 128
  float* gemb = sm + 128;      // 128
  float* memb = sm + 256;      // 128
  float* eff  = sm + 384;      // 256
  float* pmax = sm + 640;      // 256
  float* psum = sm + 896;      // 256
  float* gms  = sm + 1152;     // 2

  _Float16* h1 = (_Float16*)agg;  // reuse after convs
  _Float16* h2 = zf16;

  const int nrt = Npad >> 4;   // row tiles

  // ---------- pack weights to WMMA-B layout ----------
  auto pack = [&](const float* W, _Float16* o, int K, int M) {
    int total = K * M;
    pack_weight<<<dim3((total + 255) / 256), dim3(256), 0, stream>>>(W, o, K, M);
  };
  pack(g0W2, p_g0W2, 128, 128);
  pack(g1W1, p_g1W1, 128, 128);
  pack(g1W2, p_g1W2, 128, 128);
  pack(g2W1, p_g2W1, 128, 128);
  pack(g2W2, p_g2W2, 128, 128);
  pack(dW1,  p_d1,   128, 256);   // only top 128 rows of d_W1
  pack(dW2,  p_d2,   256, 128);

  // ---------- GIN layer 0 ----------
  feat_init<<<dim3((N + 255) / 256), dim3(256), 0, stream>>>(tss, tsr, tct, x0, agg3, N);
  scatter3<<<dim3((E + 255) / 256), dim3(256), 0, stream>>>(x0, esrc, edst, agg3, E);
  gin0_mlp1<<<dim3((N * 128 + 255) / 256), dim3(256), 0, stream>>>(x0, agg3, g0W1, g0b1, tf16, N);
  gemm16<true><<<dim3(nrt), dim3(256), 0, stream>>>(tf16, p_g0W2, g0b2, xf16, 128, 128);

  // ---------- GIN layers 1,2 ----------
  const _Float16* pw1[2] = { p_g1W1, p_g2W1 };
  const _Float16* pw2[2] = { p_g1W2, p_g2W2 };
  const float* pb1[2] = { g1b1, g2b1 };
  const float* pb2[2] = { g1b2, g2b2 };
  for (int l = 0; l < 2; ++l) {
    zero_f32<<<dim3((N * 128 + 255) / 256), dim3(256), 0, stream>>>(agg, N * 128);
    long long sth = (long long)E * 32;
    scatter128<<<dim3((unsigned)((sth + 255) / 256)), dim3(256), 0, stream>>>(xf16, esrc, edst, agg, E);
    add_cvt<<<dim3((N * 32 + 255) / 256), dim3(256), 0, stream>>>(xf16, agg, zf16, N);
    gemm16<true><<<dim3(nrt), dim3(256), 0, stream>>>(zf16, pw1[l], pb1[l], tf16, 128, 128);
    if (l == 0)
      gemm16<true ><<<dim3(nrt), dim3(256), 0, stream>>>(tf16, pw2[l], pb2[l], xf16, 128, 128);
    else
      gemm16<false><<<dim3(nrt), dim3(256), 0, stream>>>(tf16, pw2[l], pb2[l], xf16, 128, 128);
  }

  // ---------- graph / machine embeddings ----------
  zero_f32<<<dim3(1), dim3(128), 0, stream>>>(gsum, 128);
  colsum<<<dim3(256), dim3(128), 0, stream>>>(xf16, gsum, N);
  finalize_gemb<<<dim3(1), dim3(128), 0, stream>>>(gsum, gemb, (float*)d_out, N);
  machine_mlp<<<dim3(1), dim3(128), 0, stream>>>(vct, mW1, mb1, mW2, mb2, memb, (float*)d_out, N);
  eff_bias<<<dim3(1), dim3(256), 0, stream>>>(gemb, memb, dW1, db1, eff);

  // ---------- decoder (broadcast parts folded into eff bias) ----------
  gemm16<true><<<dim3(nrt), dim3(256), 0, stream>>>(xf16, p_d1, eff, h1, 128, 256);
  gemm16<true><<<dim3(nrt), dim3(256), 0, stream>>>(h1,   p_d2, db2, h2, 256, 128);
  score_kernel<<<dim3((N + 255) / 256), dim3(256), 0, stream>>>(h2, dW3, db3, tsr, scr, N);

  // ---------- masked softmax ----------
  part_max<<<dim3(256), dim3(256), 0, stream>>>(scr, pmax, N);
  final_max<<<dim3(1), dim3(256), 0, stream>>>(pmax, gms);
  part_sum<<<dim3(256), dim3(256), 0, stream>>>(scr, gms, psum, N);
  final_sum<<<dim3(1), dim3(256), 0, stream>>>(psum, gms);
  probs_kernel<<<dim3((N + 255) / 256), dim3(256), 0, stream>>>(scr, gms, (float*)d_out, N);
}
